// GAT_37031208026114
// MI455X (gfx1250) — matrix operand order repaired and checked by reading, not compiled, not run on hardware
//
#include <hip/hip_runtime.h>
#include <hip/hip_bf16.h>

#define N_NODES   100000
#define IN_DIM    512
#define HID       5
#define HEADS1    5
#define NCLS      3
#define F1        (HEADS1*HID)      // 25
#define F1PAD     32
#define NEG_SLOPE 0.2f
#define KTILES    (IN_DIM/32)       // 16

typedef __attribute__((ext_vector_type(16))) __bf16 v16bf;
typedef __attribute__((ext_vector_type(8)))  float  v8f;

// ---------- helpers ----------
// order-preserving float<->uint for atomicMax
__device__ __forceinline__ unsigned fenc(float f) {
    unsigned u = __float_as_uint(f);
    return (u & 0x80000000u) ? ~u : (u | 0x80000000u);
}
__device__ __forceinline__ float fdec(unsigned u) {
    unsigned b = (u & 0x80000000u) ? (u & 0x7FFFFFFFu) : ~u;
    return __uint_as_float(b);
}
__device__ __forceinline__ float lrelu(float x) { return x > 0.f ? x : NEG_SLOPE * x; }
__device__ __forceinline__ float logsig(float x) {
    float m = x < 0.f ? x : 0.f;
    return m - log1pf(__expf(-fabsf(x)));
}

// ---------- W1 f32 [512][25] -> bf16 pre-swizzled into WMMA B-operand layout ----------
// Linear layout: [kk (16)][tile t (2)][lane (32)][halfword j (16)]
//   K   = kk*32 + (lane>>4)*16 + j
//   col = t*16 + (lane&15)
__global__ void convW1_kernel(const float* __restrict__ W1, __bf16* __restrict__ Wb) {
    int i = blockIdx.x * blockDim.x + threadIdx.x;
    if (i >= KTILES * 2 * 32 * 16) return;
    int j    = i & 15;
    int lane = (i >> 4) & 31;
    int t    = (i >> 9) & 1;
    int kk   = i >> 10;
    int K    = kk * 32 + (lane >> 4) * 16 + j;
    int col  = t * 16 + (lane & 15);
    Wb[i] = (col < F1) ? (__bf16)W1[K * F1 + col] : (__bf16)0.f;
}

// ---------- h1 = x @ W1  (WMMA bf16 -> f32), h1 stride F1PAD ----------
__global__ void __launch_bounds__(256)
gemm1_wmma_kernel(const float* __restrict__ x, const __bf16* __restrict__ Wb,
                  float* __restrict__ h1, int nRowTiles) {
    __shared__ v16bf ldsW[KTILES * 2 * 32];           // 32 KB, already B-layout
    {   // cooperative stage of the whole swizzled W1 into LDS
        const unsigned* s = (const unsigned*)Wb;
        unsigned* d = (unsigned*)ldsW;
        for (int i = threadIdx.x; i < KTILES * 2 * 32 * 16 / 2; i += blockDim.x) d[i] = s[i];
    }
    __syncthreads();

    int warp = blockIdx.x * (blockDim.x >> 5) + (threadIdx.x >> 5);
    if (warp >= nRowTiles) return;
    int lane  = threadIdx.x & 31;
    int laneM = lane & 15;          // row within tile (A) / column (B,C)
    int laneH = lane >> 4;          // K-half selector

    const float* xrow = x + (size_t)(warp * 16 + laneM) * IN_DIM;

    v8f acc0 = {}; v8f acc1 = {};
    #pragma unroll 4
    for (int kk = 0; kk < KTILES; ++kk) {
        int k0 = kk * 32;
        __builtin_prefetch(xrow + k0 + 64, 0, 0);     // global_prefetch_b8
        // A tile 16x32 bf16: lanes 0-15 hold K {0..15}, lanes 16-31 hold K {16..31}
        v16bf a;
        #pragma unroll
        for (int j = 0; j < 8; ++j) {
            a[j]     = (__bf16)xrow[k0 + laneH * 16 + j];       // v_cvt bf16
            a[j + 8] = (__bf16)xrow[k0 + laneH * 16 + 8 + j];
        }
        // B fragments: contiguous 32B per lane (ds_load_b128 x2 each)
        v16bf b0 = ldsW[(kk * 2 + 0) * 32 + lane];
        v16bf b1 = ldsW[(kk * 2 + 1) * 32 + lane];
        acc0 = __builtin_amdgcn_wmma_f32_16x16x32_bf16(false, a, false, b0, (short)0, acc0, false, false);
        acc1 = __builtin_amdgcn_wmma_f32_16x16x32_bf16(false, a, false, b1, (short)0, acc1, false, false);
    }
    // C/D layout: VGPR r, lanes 0-15 -> M=r, lanes 16-31 -> M=r+8; N = laneM (+16 for acc1)
    int orow = warp * 16 + laneH * 8;
    #pragma unroll
    for (int r = 0; r < 8; ++r) {
        h1[(size_t)(orow + r) * F1PAD + laneM]      = acc0[r];
        h1[(size_t)(orow + r) * F1PAD + 16 + laneM] = acc1[r];
    }
}

// ---------- per-node attention logits, layer 1 ----------
__global__ void alpha1_kernel(const float* __restrict__ h1,
                              const float* __restrict__ a_src, const float* __restrict__ a_dst,
                              float* __restrict__ as1, float* __restrict__ ad1) {
    int n = blockIdx.x * blockDim.x + threadIdx.x;
    if (n >= N_NODES) return;
    const float* hv = h1 + (size_t)n * F1PAD;
    #pragma unroll
    for (int h = 0; h < HEADS1; ++h) {
        float s = 0.f, d = 0.f;
        #pragma unroll
        for (int c = 0; c < HID; ++c) {
            float v = hv[h * HID + c];
            s += v * a_src[h * HID + c];
            d += v * a_dst[h * HID + c];
        }
        as1[n * HEADS1 + h] = s;
        ad1[n * HEADS1 + h] = d;
    }
}

// ---------- layer-1 edge passes ----------
__global__ void edge1_max_kernel(const int* __restrict__ src, const int* __restrict__ dst,
                                 const float* __restrict__ as1, const float* __restrict__ ad1,
                                 unsigned* __restrict__ m1, int E, int ET) {
    int e = blockIdx.x * blockDim.x + threadIdx.x;
    if (e >= ET) return;
    int s = (e < E) ? src[e] : (e - E);
    int d = (e < E) ? dst[e] : (e - E);
    #pragma unroll
    for (int h = 0; h < HEADS1; ++h) {
        float v = lrelu(as1[s * HEADS1 + h] + ad1[d * HEADS1 + h]);
        atomicMax(&m1[d * HEADS1 + h], fenc(v));
    }
}
__global__ void edge1_sum_kernel(const int* __restrict__ src, const int* __restrict__ dst,
                                 const float* __restrict__ as1, const float* __restrict__ ad1,
                                 const unsigned* __restrict__ m1, float* __restrict__ den1,
                                 int E, int ET) {
    int e = blockIdx.x * blockDim.x + threadIdx.x;
    if (e >= ET) return;
    int s = (e < E) ? src[e] : (e - E);
    int d = (e < E) ? dst[e] : (e - E);
    #pragma unroll
    for (int h = 0; h < HEADS1; ++h) {
        float v = lrelu(as1[s * HEADS1 + h] + ad1[d * HEADS1 + h]);
        atomicAdd(&den1[d * HEADS1 + h], __expf(v - fdec(m1[d * HEADS1 + h])));
    }
}
__global__ void edge1_agg_kernel(const int* __restrict__ src, const int* __restrict__ dst,
                                 const float* __restrict__ as1, const float* __restrict__ ad1,
                                 const unsigned* __restrict__ m1, const float* __restrict__ den1,
                                 const float* __restrict__ h1, float* __restrict__ out1,
                                 int E, int ET) {
    int e = blockIdx.x * blockDim.x + threadIdx.x;
    if (e >= ET) return;
    int s = (e < E) ? src[e] : (e - E);
    int d = (e < E) ? dst[e] : (e - E);
    const float* hs = h1 + (size_t)s * F1PAD;
    float* od = out1 + (size_t)d * F1PAD;
    #pragma unroll
    for (int h = 0; h < HEADS1; ++h) {
        float v = lrelu(as1[s * HEADS1 + h] + ad1[d * HEADS1 + h]);
        float alpha = __expf(v - fdec(m1[d * HEADS1 + h])) / den1[d * HEADS1 + h];
        #pragma unroll
        for (int c = 0; c < HID; ++c)
            atomicAdd(&od[h * HID + c], alpha * hs[h * HID + c]);
    }
}

// ---------- layer-1 finalize (bias + ELU) + layer-2 projection + logits ----------
__global__ void finalize1_kernel(const float* __restrict__ out1, const float* __restrict__ bias1,
                                 const float* __restrict__ W2,
                                 const float* __restrict__ a_src2, const float* __restrict__ a_dst2,
                                 float* __restrict__ h2, float* __restrict__ as2, float* __restrict__ ad2) {
    int n = blockIdx.x * blockDim.x + threadIdx.x;
    if (n >= N_NODES) return;
    const float* ov = out1 + (size_t)n * F1PAD;
    float o[NCLS] = {0.f, 0.f, 0.f};
    #pragma unroll
    for (int j = 0; j < F1; ++j) {
        float t = ov[j] + bias1[j];
        float hv = t > 0.f ? t : (__expf(t) - 1.f);       // ELU
        #pragma unroll
        for (int c = 0; c < NCLS; ++c) o[c] += hv * W2[j * NCLS + c];
    }
    float s = 0.f, d = 0.f;
    #pragma unroll
    for (int c = 0; c < NCLS; ++c) {
        h2[n * 4 + c] = o[c];
        s += o[c] * a_src2[c];
        d += o[c] * a_dst2[c];
    }
    as2[n] = s; ad2[n] = d;
}

// ---------- layer-2 edge passes (1 head, 3 channels) ----------
__global__ void edge2_max_kernel(const int* __restrict__ src, const int* __restrict__ dst,
                                 const float* __restrict__ as2, const float* __restrict__ ad2,
                                 unsigned* __restrict__ m2, int E, int ET) {
    int e = blockIdx.x * blockDim.x + threadIdx.x;
    if (e >= ET) return;
    int s = (e < E) ? src[e] : (e - E);
    int d = (e < E) ? dst[e] : (e - E);
    atomicMax(&m2[d], fenc(lrelu(as2[s] + ad2[d])));
}
__global__ void edge2_sum_kernel(const int* __restrict__ src, const int* __restrict__ dst,
                                 const float* __restrict__ as2, const float* __restrict__ ad2,
                                 const unsigned* __restrict__ m2, float* __restrict__ den2,
                                 int E, int ET) {
    int e = blockIdx.x * blockDim.x + threadIdx.x;
    if (e >= ET) return;
    int s = (e < E) ? src[e] : (e - E);
    int d = (e < E) ? dst[e] : (e - E);
    float v = lrelu(as2[s] + ad2[d]);
    atomicAdd(&den2[d], __expf(v - fdec(m2[d])));
}
__global__ void edge2_agg_kernel(const int* __restrict__ src, const int* __restrict__ dst,
                                 const float* __restrict__ as2, const float* __restrict__ ad2,
                                 const unsigned* __restrict__ m2, const float* __restrict__ den2,
                                 const float* __restrict__ h2, float* __restrict__ out2,
                                 int E, int ET) {
    int e = blockIdx.x * blockDim.x + threadIdx.x;
    if (e >= ET) return;
    int s = (e < E) ? src[e] : (e - E);
    int d = (e < E) ? dst[e] : (e - E);
    float v = lrelu(as2[s] + ad2[d]);
    float alpha = __expf(v - fdec(m2[d])) / den2[d];
    #pragma unroll
    for (int c = 0; c < NCLS; ++c)
        atomicAdd(&out2[d * 4 + c], alpha * h2[s * 4 + c]);
}

// ---------- epilogue: + bias2, log_sigmoid ----------
__global__ void final_kernel(const float* __restrict__ out2, const float* __restrict__ bias2,
                             float* __restrict__ out) {
    int n = blockIdx.x * blockDim.x + threadIdx.x;
    if (n >= N_NODES) return;
    #pragma unroll
    for (int c = 0; c < NCLS; ++c)
        out[n * NCLS + c] = logsig(out2[n * 4 + c] + bias2[c]);
}

extern "C" void kernel_launch(void* const* d_in, const int* in_sizes, int n_in,
                              void* d_out, int out_size, void* d_ws, size_t ws_size,
                              hipStream_t stream) {
    const float* x        = (const float*)d_in[0];
    const int*   ei       = (const int*)d_in[1];
    const float* W1       = (const float*)d_in[2];
    const float* att_src1 = (const float*)d_in[3];
    const float* att_dst1 = (const float*)d_in[4];
    const float* bias1    = (const float*)d_in[5];
    const float* W2       = (const float*)d_in[6];
    const float* att_src2 = (const float*)d_in[7];
    const float* att_dst2 = (const float*)d_in[8];
    const float* bias2    = (const float*)d_in[9];
    float* out = (float*)d_out;

    const int E  = in_sizes[1] / 2;     // edge_index is [2, E]
    const int ET = E + N_NODES;         // + self loops
    const int* src = ei;
    const int* dst = ei + E;

    // ---- workspace layout (floats) ----
    const size_t N = N_NODES;
    float*    h1   = (float*)d_ws;                       // N*32
    float*    as1  = h1 + N * F1PAD;                     // N*5
    float*    ad1  = as1 + N * HEADS1;                   // N*5
    unsigned* m1   = (unsigned*)(ad1 + N * HEADS1);      // N*5   } zero
    float*    den1 = (float*)(m1 + N * HEADS1);          // N*5   } zero
    float*    out1 = den1 + N * HEADS1;                  // N*32  } zero
    float*    h2   = out1 + N * F1PAD;                   // N*4
    float*    as2  = h2 + N * 4;                         // N
    float*    ad2  = as2 + N;                            // N
    unsigned* m2   = (unsigned*)(ad2 + N);               // N     } zero
    float*    den2 = (float*)(m2 + N);                   // N     } zero
    float*    out2 = den2 + N;                           // N*4   } zero
    __bf16*   Wb   = (__bf16*)(out2 + N * 4);            // 512*32 bf16 (swizzled)

    // zero the atomic accumulation regions (contiguous blocks); fenc identity = 0
    hipMemsetAsync(m1, 0, N * (HEADS1 * 2 + F1PAD) * sizeof(float), stream);   // m1+den1+out1
    hipMemsetAsync(m2, 0, N * 6 * sizeof(float), stream);                       // m2+den2+out2

    const int B = 256;
    convW1_kernel<<<(KTILES * 2 * 32 * 16 + B - 1) / B, B, 0, stream>>>(W1, Wb);

    const int nRowTiles = N_NODES / 16;                  // 6250 exact
    const int warpsPerBlk = B / 32;
    gemm1_wmma_kernel<<<(nRowTiles + warpsPerBlk - 1) / warpsPerBlk, B, 0, stream>>>(x, Wb, h1, nRowTiles);

    const int nb = (N_NODES + B - 1) / B;
    const int eb = (ET + B - 1) / B;

    alpha1_kernel<<<nb, B, 0, stream>>>(h1, att_src1, att_dst1, as1, ad1);
    edge1_max_kernel<<<eb, B, 0, stream>>>(src, dst, as1, ad1, m1, E, ET);
    edge1_sum_kernel<<<eb, B, 0, stream>>>(src, dst, as1, ad1, m1, den1, E, ET);
    edge1_agg_kernel<<<eb, B, 0, stream>>>(src, dst, as1, ad1, m1, den1, h1, out1, E, ET);

    finalize1_kernel<<<nb, B, 0, stream>>>(out1, bias1, W2, att_src2, att_dst2, h2, as2, ad2);
    edge2_max_kernel<<<eb, B, 0, stream>>>(src, dst, as2, ad2, m2, E, ET);
    edge2_sum_kernel<<<eb, B, 0, stream>>>(src, dst, as2, ad2, m2, den2, E, ET);
    edge2_agg_kernel<<<eb, B, 0, stream>>>(src, dst, as2, ad2, m2, den2, h2, out2, E, ET);

    final_kernel<<<nb, B, 0, stream>>>(out2, bias2, out);
}
